// Rhn_21225728377122
// MI455X (gfx1250) — matrix operand order repaired and checked by reading, not compiled, not run on hardware
//
#include <hip/hip_runtime.h>
#include <hip/hip_bf16.h>
#include <math.h>

// Recurrent Highway Network on MI455X (gfx1250).
//  - Persistent kernel, 32 workgroups x 256 threads (8 wave32) = 256 waves.
//  - Each wave owns one 16x16 output tile (m-tile, n-strip) for ALL 3 gates ->
//    highway update is wave-local. bf16 WMMA 16x16x32, f32 accumulate.
//  - Input projections fused into layer 0 as extra K ([s | x_t], K=2048).
//  - Weights pre-packed bf16, gate-interleaved: per (l,ns,ks) the 3 gate
//    fragments are contiguous (3KB) -> sequential weight stream, small
//    ioffset immediates, L2-resident (60MB).
//  - Device-wide software barrier between the 2560 dependent layer-GEMMs.

typedef __attribute__((ext_vector_type(16))) __bf16 v16bf;
typedef __attribute__((ext_vector_type(8)))  float  v8f;

#define H      1024
#define BATCH  64
#define TSTEPS 512
#define NL     5
#define NWG    32
#define WPW    8              // waves per workgroup

// ws layout (bytes)
#define WS_BAR_OFF   0
#define WS_S0_OFF    1024
#define WS_S1_OFF    (WS_S0_OFF + BATCH*H*4)
#define WS_WPACK_OFF (WS_S1_OFF + BATCH*H*4)
// packed blocks: index = (((l*64 + ns)*64 + ks)*3 + g) * 512 bf16 elems
#define KS_STRIDE    (3*512)            // elems per ks step (3 gates contiguous)
#define WPACK_ELEMS  (5*64*64*3*512)    // 5 layers * 64 ns * 64 ks * 3 gates * 512

// ---------------------------------------------------------------- init ------
__global__ void rhn_init(const float* __restrict__ s0, float* __restrict__ S,
                         unsigned* __restrict__ bar) {
  int i = blockIdx.x * blockDim.x + threadIdx.x;
  if (i < 2) bar[i] = 0u;               // ws is poisoned 0xAA -> must zero barrier
  if (i < BATCH * H) S[i] = s0[i];
}

// ------------------------------------------------------- weight packing -----
// One WMMA B fragment = 32 lanes x 16 bf16 = 512 elems. B layout (16-bit,
// 32x16): lanes 0-15 = column n, K {0..7,16..23}; lanes 16-31 = column n, K {8..15,24..31}.
// ks 0..31  -> recurrent weight r_*w[l];  ks 32..63 -> input weight w_* (l==0).
__global__ void rhn_pack(const float* __restrict__ w_h, const float* __restrict__ w_t,
                         const float* __restrict__ w_c, const float* __restrict__ r_hw,
                         const float* __restrict__ r_tw, const float* __restrict__ r_cw,
                         __bf16* __restrict__ wpack) {
  size_t idx = (size_t)blockIdx.x * blockDim.x + threadIdx.x;
  if (idx >= (size_t)WPACK_ELEMS) return;
  int e  = (int)(idx & 511);
  size_t blk = idx >> 9;
  int g  = (int)(blk % 3); blk /= 3;
  int ks = (int)(blk & 63); blk >>= 6;
  int ns = (int)(blk & 63); blk >>= 6;
  int l  = (int)blk;
  int lane = e >> 4, j = e & 15;
  int n = ns * 16 + (lane & 15);
  int k_local = ((j < 8) ? j : (8 + j)) + ((lane < 16) ? 0 : 8);
  float v = 0.0f;
  if (ks < 32) {
    const float* R = (g == 0) ? r_hw : ((g == 1) ? r_tw : r_cw);
    v = R[(size_t)l * H * H + (size_t)(ks * 32 + k_local) * H + n];
  } else if (l == 0) {
    const float* W = (g == 0) ? w_h : ((g == 1) ? w_t : w_c);
    v = W[(size_t)((ks - 32) * 32 + k_local) * H + n];
  }
  wpack[idx] = (__bf16)v;
}

// ------------------------------------------------------------- helpers ------
// A fragment (16-bit A 16x32): lane&15 = row M; lanes 0-15 hold K {0..7,16..23},
// lanes 16-31 hold K {8..15,24..31}. p already includes row*stride + khalf.
__device__ inline v16bf cvt_a_frag(const float* p) {
  float4 a0 = *(const float4*)(p);
  float4 a1 = *(const float4*)(p + 4);
  float4 a2 = *(const float4*)(p + 16);
  float4 a3 = *(const float4*)(p + 20);
  v16bf r;
  r[0]=(__bf16)a0.x;  r[1]=(__bf16)a0.y;  r[2]=(__bf16)a0.z;  r[3]=(__bf16)a0.w;
  r[4]=(__bf16)a1.x;  r[5]=(__bf16)a1.y;  r[6]=(__bf16)a1.z;  r[7]=(__bf16)a1.w;
  r[8]=(__bf16)a2.x;  r[9]=(__bf16)a2.y;  r[10]=(__bf16)a2.z; r[11]=(__bf16)a2.w;
  r[12]=(__bf16)a3.x; r[13]=(__bf16)a3.y; r[14]=(__bf16)a3.z; r[15]=(__bf16)a3.w;
  return r;
}

// 3 WMMAs sharing A (reuse_a hint on the first two identical instructions).
__device__ inline void wmma3(const v16bf& a, const __bf16* wp,
                             v8f& accH, v8f& accT, v8f& accC) {
  v16bf bh = *(const v16bf*)(wp);
  v16bf bt = *(const v16bf*)(wp + 512);
  v16bf bc = *(const v16bf*)(wp + 1024);
  accH = __builtin_amdgcn_wmma_f32_16x16x32_bf16(false, a, false, bh, (short)0, accH, true,  false);
  accT = __builtin_amdgcn_wmma_f32_16x16x32_bf16(false, a, false, bt, (short)0, accT, true,  false);
  accC = __builtin_amdgcn_wmma_f32_16x16x32_bf16(false, a, false, bc, (short)0, accC, false, false);
}

// Device-wide barrier. All NWG workgroups are co-resident (32 WGs, 256 thr, 0 LDS).
__device__ inline void grid_barrier(unsigned* bar) {
  __syncthreads();
  if (threadIdx.x == 0) {
    __threadfence();
    unsigned gen  = __hip_atomic_load(bar + 1, __ATOMIC_RELAXED, __HIP_MEMORY_SCOPE_AGENT);
    unsigned prev = __hip_atomic_fetch_add(bar, 1u, __ATOMIC_ACQ_REL, __HIP_MEMORY_SCOPE_AGENT);
    if (prev == NWG - 1) {
      __hip_atomic_store(bar, 0u, __ATOMIC_RELAXED, __HIP_MEMORY_SCOPE_AGENT);
      __hip_atomic_store(bar + 1, gen + 1u, __ATOMIC_RELEASE, __HIP_MEMORY_SCOPE_AGENT);
    } else {
      while (__hip_atomic_load(bar + 1, __ATOMIC_ACQUIRE, __HIP_MEMORY_SCOPE_AGENT) == gen) {
        __builtin_amdgcn_s_sleep(2);
      }
    }
    __threadfence();
  }
  __syncthreads();
}

// --------------------------------------------------------- main kernel ------
__global__ void __launch_bounds__(256) rhn_recurrent(
    const float* __restrict__ x, const __bf16* __restrict__ wpack,
    const float* __restrict__ r_hb, const float* __restrict__ r_tb,
    const float* __restrict__ r_cb,
    float* Sa, float* Sb,
    float* __restrict__ out, float* __restrict__ sfinal,
    unsigned* bar) {
  const int lane  = threadIdx.x & 31;
  const int gw    = blockIdx.x * WPW + (threadIdx.x >> 5); // 0..255
  const int mt    = gw & 3;                                // m-tile (batch rows)
  const int ns    = gw >> 2;                               // n-strip 0..63
  const int nloc  = lane & 15;
  const int n     = ns * 16 + nloc;
  const int mhalf = (lane >> 4) << 3;                      // C/D: lanes 16-31 -> M+8
  const int arow  = mt * 16 + nloc;                        // A row this lane reads
  const int khalf = (lane >> 4) << 3;                      // A: lanes 16-31 -> K+8

  float* Scur = Sa;
  float* Snxt = Sb;

  for (int t = 0; t < TSTEPS; ++t) {
    // x_t A-fragment base for this lane (row stride = TSTEPS*H)
    const float* pX = x + (size_t)arow * (TSTEPS * H) + (size_t)t * H + khalf;
    for (int l = 0; l < NL; ++l) {
      v8f accH = {0.f,0.f,0.f,0.f,0.f,0.f,0.f,0.f};
      v8f accT = accH;
      v8f accC = accH;
      // per-lane base pointers (loop-invariant; per-ks offsets are immediates)
      const __bf16* pB = wpack + ((size_t)(l * 64 + ns) * 64) * KS_STRIDE + (size_t)lane * 16;
      const float*  pA = Scur + (size_t)arow * H + khalf;

      // K = hidden state (all layers), branch-free
#pragma unroll 4
      for (int ks = 0; ks < 32; ++ks) {
        v16bf a = cvt_a_frag(pA + ks * 32);
        wmma3(a, pB + (size_t)ks * KS_STRIDE, accH, accT, accC);
      }
      // K = x_t (layer 0 only): fused input projection
      if (l == 0) {
#pragma unroll 4
        for (int ks = 0; ks < 32; ++ks) {
          v16bf a = cvt_a_frag(pX + ks * 32);
          wmma3(a, pB + (size_t)(32 + ks) * KS_STRIDE, accH, accT, accC);
        }
      }

      // highway update epilogue (wave-local: this wave owns all 3 gates)
      const float bhv = r_hb[l * H + n];
      const float btv = r_tb[l * H + n];
      const float bcv = r_cb[l * H + n];
#pragma unroll
      for (int r = 0; r < 8; ++r) {
        const int m = mt * 16 + mhalf + r;               // C/D: vgpr r -> M=r / r+8
        const float sold = Scur[(size_t)m * H + n];
        const float hg = tanhf(accH[r] + bhv);
        const float tg = 1.0f / (1.0f + __expf(-(accT[r] + btv)));
        const float cg = 1.0f / (1.0f + __expf(-(accC[r] + bcv)));
        const float sn = hg * tg + sold * cg;
        Snxt[(size_t)m * H + n] = sn;
        if (l == NL - 1) {
          out[((size_t)m * TSTEPS + t) * H + n] = sn;
          if (t == TSTEPS - 1) sfinal[(size_t)m * H + n] = sn;
        }
      }
      grid_barrier(bar);
      float* tmp = Scur; Scur = Snxt; Snxt = tmp;        // ping-pong state
    }
  }
}

// ------------------------------------------------------------- launch -------
extern "C" void kernel_launch(void* const* d_in, const int* in_sizes, int n_in,
                              void* d_out, int out_size, void* d_ws, size_t ws_size,
                              hipStream_t stream) {
  const float* x    = (const float*)d_in[0];
  const float* s0   = (const float*)d_in[1];
  const float* w_h  = (const float*)d_in[2];
  const float* w_t  = (const float*)d_in[3];
  const float* w_c  = (const float*)d_in[4];
  const float* r_hw = (const float*)d_in[5];
  const float* r_tw = (const float*)d_in[6];
  const float* r_cw = (const float*)d_in[7];
  const float* r_hb = (const float*)d_in[8];
  const float* r_tb = (const float*)d_in[9];
  const float* r_cb = (const float*)d_in[10];

  char* ws = (char*)d_ws;
  unsigned* bar  = (unsigned*)(ws + WS_BAR_OFF);
  float*   Sa    = (float*)(ws + WS_S0_OFF);
  float*   Sb    = (float*)(ws + WS_S1_OFF);
  __bf16*  wpack = (__bf16*)(ws + WS_WPACK_OFF);

  float* out    = (float*)d_out;
  float* sfinal = out + (size_t)BATCH * TSTEPS * H;

  rhn_init<<<(BATCH * H + 255) / 256, 256, 0, stream>>>(s0, Sa, bar);
  rhn_pack<<<(WPACK_ELEMS + 255) / 256, 256, 0, stream>>>(w_h, w_t, w_c,
                                                          r_hw, r_tw, r_cw, wpack);
  rhn_recurrent<<<NWG, 256, 0, stream>>>(x, wpack, r_hb, r_tb, r_cb,
                                         Sa, Sb, out, sfinal, bar);
}
